// LinearAttention_83339545411776
// MI455X (gfx1250) — matrix operand order repaired and checked by reading, not compiled, not run on hardware
//
#include <hip/hip_runtime.h>
#include <hip/hip_bf16.h>
#include <stdint.h>

typedef __attribute__((ext_vector_type(16))) __bf16 v16bf;
typedef __attribute__((ext_vector_type(8)))  float  v8f;
typedef int v4i_gcc __attribute__((vector_size(16)));  // matches builtin param type

constexpr int kB = 4, kS = 4096, kD = 1024, kH = 16, kHD = 64, kE = 3072;

// native bf16 convert (v_cvt_bf16_f32 / v_cvt_pk_bf16_f32 on gfx1250)
__device__ __forceinline__ unsigned short f2bf(float f) {
  union { __bf16 h; unsigned short u; } c;
  c.h = (__bf16)f;
  return c.u;
}
__device__ __forceinline__ float bf2f(unsigned short h) {
  union { unsigned u; float f; } c; c.u = ((unsigned)h) << 16; return c.f;
}
__device__ __forceinline__ v8f zero_v8f() {
  v8f z;
#pragma unroll
  for (int i = 0; i < 8; ++i) z[i] = 0.f;
  return z;
}

union Frag {
  v16bf v;
  unsigned short u[16];
  uint4 q[2];
};

// per-lane async global->LDS 16B copy (ASYNCcnt tracked)
__device__ __forceinline__ void async_copy_b128(const void* g, void* l) {
  __builtin_amdgcn_global_load_async_to_lds_b128(
      (__attribute__((address_space(1))) v4i_gcc*)(void*)g,
      (__attribute__((address_space(3))) v4i_gcc*)l, 0, 0);
}
__device__ __forceinline__ void wait_async0() {
  asm volatile("s_wait_asynccnt 0" ::: "memory");
}
__device__ __forceinline__ void wait_async2() {  // allow 2 (next tile) in flight
  asm volatile("s_wait_asynccnt 2" ::: "memory");
}

// ---------------------------------------------------------------- kernel 0
// vectorized fp32 -> bf16 (n4 = n/4)
__global__ void cvt_f32_bf16(const float* __restrict__ src,
                             unsigned short* __restrict__ dst, int n4) {
  int i = blockIdx.x * blockDim.x + threadIdx.x;
  for (; i < n4; i += gridDim.x * blockDim.x) {
    float4 f = reinterpret_cast<const float4*>(src)[i];
    union { unsigned short u[4]; uint2 q; } t;
    t.u[0] = f2bf(f.x); t.u[1] = f2bf(f.y);
    t.u[2] = f2bf(f.z); t.u[3] = f2bf(f.w);
    reinterpret_cast<uint2*>(dst)[i] = t.q;
  }
}

// ---------------------------------------------------------------- kernel 1
// qkv = x @ w_qkv^T + b_qkv ; elu+1 on q,k ; scatter:
//   q  -> qb  [bh][s][64]   (row major)
//   k  -> ktb [bh][64][s]   (transposed: s contiguous)
//   v  -> vtb [bh][64][s]
__global__ __launch_bounds__(256) void gemm_qkv(
    const unsigned short* __restrict__ xb, const unsigned short* __restrict__ wbf,
    const float* __restrict__ bias,
    unsigned short* __restrict__ qb, unsigned short* __restrict__ ktb,
    unsigned short* __restrict__ vtb) {
  __shared__ __align__(16) unsigned short smem[128 * 136];  // 2x A-stage | out-stage alias
  unsigned short* smemO = smem;  // epilogue: 128 x 136 bf16

  const int tid = threadIdx.x;
  const int lane = tid & 31, wave = tid >> 5;
  const int wm = wave & 1, wn = wave >> 1;
  const int hl = lane >> 4, lc = lane & 15;
  const int mbase = blockIdx.x * 128;
  const int nbase = blockIdx.y * 128;

  v8f acc[4][2];
#pragma unroll
  for (int i = 0; i < 4; ++i)
#pragma unroll
    for (int j = 0; j < 2; ++j) acc[i][j] = zero_v8f();

  // async-stage one 128x32 bf16 tile (512 x 16B chunks, 2 per thread)
  auto issue_tile = [&](int kk, unsigned short* buf) {
#pragma unroll
    for (int c = 0; c < 2; ++c) {
      int chunk = tid + c * 256;
      int row = chunk >> 2, qd = chunk & 3;
      const unsigned short* g = xb + (size_t)(mbase + row) * kD + kk + qd * 8;
      async_copy_b128(g, buf + row * 40 + qd * 8);
    }
  };

  auto compute = [&](const unsigned short* smemA, int kk) {
    Frag bfr[2];
#pragma unroll
    for (int nsub = 0; nsub < 2; ++nsub) {  // B direct from global (K contiguous)
      int col = nbase + wn * 32 + nsub * 16 + lc;
      const unsigned short* p = wbf + (size_t)col * kD + kk + hl * 8;
      bfr[nsub].q[0] = *reinterpret_cast<const uint4*>(p);
      bfr[nsub].q[1] = *reinterpret_cast<const uint4*>(p + 16);
    }
#pragma unroll
    for (int msub = 0; msub < 4; ++msub) {
      Frag afr;
      const unsigned short* ar = smemA + (wm * 64 + msub * 16 + lc) * 40;
      afr.q[0] = *reinterpret_cast<const uint4*>(ar + hl * 8);
      afr.q[1] = *reinterpret_cast<const uint4*>(ar + 16 + hl * 8);
#pragma unroll
      for (int nsub = 0; nsub < 2; ++nsub) {
        acc[msub][nsub] = __builtin_amdgcn_wmma_f32_16x16x32_bf16(
            false, afr.v, false, bfr[nsub].v, (short)0, acc[msub][nsub], false, false);
      }
    }
  };

  unsigned short* buf0 = smem;          // static addresses: keep LDS inference clean
  unsigned short* buf1 = smem + 5120;   // 128 x 40 each

  issue_tile(0, buf0);
  for (int i2 = 0; i2 < 16; ++i2) {  // 2 k-chunks per iteration, ping-pong buffers
    const int kk0 = i2 * 64;
    // phase A: compute buf0 @ kk0, prefetch kk0+32 into buf1 (always in range)
    issue_tile(kk0 + 32, buf1);
    wait_async2();
    __syncthreads();
    compute(buf0, kk0);
    __syncthreads();
    // phase B: compute buf1 @ kk0+32, prefetch kk0+64 into buf0
    if (i2 + 1 < 16) {
      issue_tile(kk0 + 64, buf0);
      wait_async2();
    } else {
      wait_async0();
    }
    __syncthreads();
    compute(buf1, kk0 + 32);
    __syncthreads();
  }

  // epilogue: +bias, elu+1 for q/k, bf16, stage full 128x128 tile to LDS
#pragma unroll
  for (int msub = 0; msub < 4; ++msub)
#pragma unroll
    for (int nsub = 0; nsub < 2; ++nsub) {
      int ncol = wn * 32 + nsub * 16 + lc;
      int ngl = nbase + ncol;
      float bv = bias[ngl];
      bool feat = (ngl >> 10) < 2;  // q or k get elu()+1
#pragma unroll
      for (int r = 0; r < 8; ++r) {
        int mrow = wm * 64 + msub * 16 + hl * 8 + r;
        float v = acc[msub][nsub][r] + bv;
        if (feat) v = (v > 0.f) ? (v + 1.f) : __expf(v);
        smemO[mrow * 136 + ncol] = f2bf(v);
      }
    }
  __syncthreads();

  if (blockIdx.y < 8) {  // q region: row-major coalesced store
    int row = tid >> 1, h2 = tid & 1;
    int m = mbase + row, b = m >> 12, s = m & 4095;
    int head = (nbase + h2 * 64) >> 6;
    unsigned short* dst = qb + (((size_t)(b * kH + head)) * kS + s) * kHD;
    const uint4* sp = reinterpret_cast<const uint4*>(smemO + row * 136 + h2 * 64);
    uint4* dq = reinterpret_cast<uint4*>(dst);
#pragma unroll
    for (int i = 0; i < 8; ++i) dq[i] = sp[i];
  } else {  // k / v region: transpose via LDS column reads, s-contiguous store
    int col = tid >> 1, h2 = tid & 1;
    int ngl = nbase + col;
    int which = ngl >> 10;  // 1=k, 2=v (tile is homogeneous)
    int head = (ngl >> 6) & 15, d = ngl & 63;
    int b = mbase >> 12, s0 = (mbase & 4095) + h2 * 64;
    union { uint4 q[8]; unsigned short u[64]; } t;
#pragma unroll
    for (int r = 0; r < 64; ++r) t.u[r] = smemO[(h2 * 64 + r) * 136 + col];
    unsigned short* base = (which == 1) ? ktb : vtb;
    unsigned short* dst = base + (((size_t)(b * kH + head)) * kHD + d) * kS + s0;
    uint4* dq = reinterpret_cast<uint4*>(dst);
#pragma unroll
    for (int i = 0; i < 8; ++i) dq[i] = t.q[i];
  }
}

// ---------------------------------------------------------------- kernel 2
// kv[bh] = K^T V (64x64), stored transposed bf16 [bh][dv][dk]; inv_ksum per feature
__global__ __launch_bounds__(128) void kv_state(
    const unsigned short* __restrict__ ktb, const unsigned short* __restrict__ vtb,
    unsigned short* __restrict__ kvb, float* __restrict__ invks) {
  const int bh = blockIdx.x;
  const int tid = threadIdx.x, lane = tid & 31, wave = tid >> 5;
  const int hl = lane >> 4, lc = lane & 15;

  v8f acc[4];
#pragma unroll
  for (int i = 0; i < 4; ++i) acc[i] = zero_v8f();

  const unsigned short* krow = ktb + ((size_t)bh * kHD + wave * 16 + lc) * kS;
  for (int s0 = 0; s0 < kS; s0 += 32) {
    Frag afr;  // A = K^T (16 dk rows x 32 s), s contiguous per lane
    afr.q[0] = *reinterpret_cast<const uint4*>(krow + s0 + hl * 8);
    afr.q[1] = *reinterpret_cast<const uint4*>(krow + s0 + 16 + hl * 8);
#pragma unroll
    for (int nsub = 0; nsub < 4; ++nsub) {
      Frag bfr;  // B = V (32 s x 16 dv), s contiguous per lane
      const unsigned short* vp = vtb + ((size_t)bh * kHD + nsub * 16 + lc) * kS + s0 + hl * 8;
      bfr.q[0] = *reinterpret_cast<const uint4*>(vp);
      bfr.q[1] = *reinterpret_cast<const uint4*>(vp + 16);
      acc[nsub] = __builtin_amdgcn_wmma_f32_16x16x32_bf16(
          false, afr.v, false, bfr.v, (short)0, acc[nsub], false, false);
    }
  }
#pragma unroll
  for (int nsub = 0; nsub < 4; ++nsub) {  // C lane: col dv fixed, 8 dk rows -> contiguous
    int dv = nsub * 16 + lc;
    int dk0 = wave * 16 + hl * 8;
    union { uint4 q; unsigned short u[8]; } t;
#pragma unroll
    for (int r = 0; r < 8; ++r) t.u[r] = f2bf(acc[nsub][r]);
    *reinterpret_cast<uint4*>(kvb + ((size_t)bh * kHD + dv) * kHD + dk0) = t.q;
  }
  if (tid < kHD) {  // per-feature 1/(sum_s k + eps)
    const unsigned short* kr = ktb + ((size_t)bh * kHD + tid) * kS;
    float sum = 0.f;
    for (int s = 0; s < kS; s += 8) {
      uint4 u = *reinterpret_cast<const uint4*>(kr + s);
      sum += bf2f((unsigned short)(u.x & 0xffff)) + bf2f((unsigned short)(u.x >> 16))
           + bf2f((unsigned short)(u.y & 0xffff)) + bf2f((unsigned short)(u.y >> 16))
           + bf2f((unsigned short)(u.z & 0xffff)) + bf2f((unsigned short)(u.z >> 16))
           + bf2f((unsigned short)(u.w & 0xffff)) + bf2f((unsigned short)(u.w >> 16));
    }
    invks[bh * kHD + tid] = 1.0f / (sum + 1e-6f);
  }
}

// ---------------------------------------------------------------- kernel 3
// out_head = (Q @ kv) * inv_ksum[dv] -> attn [b][s][h*64+dv] bf16
__global__ __launch_bounds__(256) void attn_apply(
    const unsigned short* __restrict__ qb, const unsigned short* __restrict__ kvb,
    const float* __restrict__ invks, unsigned short* __restrict__ attn) {
  __shared__ __align__(16) unsigned short smem3[128 * 72];
  const int bh = blockIdx.y;
  const int mbase = blockIdx.x * 128;
  const int tid = threadIdx.x, lane = tid & 31, wave = tid >> 5;
  const int hl = lane >> 4, lc = lane & 15;

  v8f acc[4];
#pragma unroll
  for (int i = 0; i < 4; ++i) acc[i] = zero_v8f();

  const unsigned short* qrow = qb + ((size_t)bh * kS + mbase + wave * 16 + lc) * kHD;
#pragma unroll
  for (int kk = 0; kk < kHD; kk += 32) {
    Frag afr;
    afr.q[0] = *reinterpret_cast<const uint4*>(qrow + kk + hl * 8);
    afr.q[1] = *reinterpret_cast<const uint4*>(qrow + kk + 16 + hl * 8);
#pragma unroll
    for (int nsub = 0; nsub < 4; ++nsub) {
      Frag bfr;
      const unsigned short* kvp = kvb + ((size_t)bh * kHD + nsub * 16 + lc) * kHD + kk + hl * 8;
      bfr.q[0] = *reinterpret_cast<const uint4*>(kvp);
      bfr.q[1] = *reinterpret_cast<const uint4*>(kvp + 16);
      acc[nsub] = __builtin_amdgcn_wmma_f32_16x16x32_bf16(
          false, afr.v, false, bfr.v, (short)0, acc[nsub], false, false);
    }
  }
#pragma unroll
  for (int nsub = 0; nsub < 4; ++nsub) {
    int dv = nsub * 16 + lc;
    float inv = invks[bh * kHD + dv];
#pragma unroll
    for (int r = 0; r < 8; ++r) {
      int mrow = wave * 16 + hl * 8 + r;
      smem3[mrow * 72 + dv] = f2bf(acc[nsub][r] * inv);
    }
  }
  __syncthreads();
  {
    int row = tid >> 1, h2 = tid & 1;
    int b = bh >> 4, head = bh & 15, s = mbase + row;
    unsigned short* dst = attn + ((size_t)b * kS + s) * kD + head * kHD + h2 * 32;
    const uint4* sp = reinterpret_cast<const uint4*>(smem3 + row * 72 + h2 * 32);
    uint4* dq = reinterpret_cast<uint4*>(dst);
    dq[0] = sp[0]; dq[1] = sp[1]; dq[2] = sp[2]; dq[3] = sp[3];
  }
}

// ---------------------------------------------------------------- kernel 4
// final = attn @ w_out^T + b_out (fp32 out)
__global__ __launch_bounds__(256) void gemm_out(
    const unsigned short* __restrict__ attn, const unsigned short* __restrict__ wbf,
    const float* __restrict__ bias, float* __restrict__ out) {
  __shared__ __align__(16) unsigned char smemraw[128 * 68 * 4];  // 2x A-stage | f32 epilogue
  float* smemF = (float*)smemraw;  // 128 x 68 f32 (epilogue)

  const int tid = threadIdx.x, lane = tid & 31, wave = tid >> 5;
  const int wm = wave & 1, wn = wave >> 1;
  const int hl = lane >> 4, lc = lane & 15;
  const int mbase = blockIdx.x * 128, nbase = blockIdx.y * 128;

  v8f acc[4][2];
#pragma unroll
  for (int i = 0; i < 4; ++i)
#pragma unroll
    for (int j = 0; j < 2; ++j) acc[i][j] = zero_v8f();

  auto issue_tile = [&](int kk, unsigned short* buf) {
#pragma unroll
    for (int c = 0; c < 2; ++c) {
      int chunk = tid + c * 256;
      int row = chunk >> 2, qd = chunk & 3;
      const unsigned short* g = attn + (size_t)(mbase + row) * kD + kk + qd * 8;
      async_copy_b128(g, buf + row * 40 + qd * 8);
    }
  };

  auto compute = [&](const unsigned short* smemA, int kk) {
    Frag bfr[2];
#pragma unroll
    for (int nsub = 0; nsub < 2; ++nsub) {
      int col = nbase + wn * 32 + nsub * 16 + lc;
      const unsigned short* p = wbf + (size_t)col * kD + kk + hl * 8;
      bfr[nsub].q[0] = *reinterpret_cast<const uint4*>(p);
      bfr[nsub].q[1] = *reinterpret_cast<const uint4*>(p + 16);
    }
#pragma unroll
    for (int msub = 0; msub < 4; ++msub) {
      Frag afr;
      const unsigned short* ar = smemA + (wm * 64 + msub * 16 + lc) * 40;
      afr.q[0] = *reinterpret_cast<const uint4*>(ar + hl * 8);
      afr.q[1] = *reinterpret_cast<const uint4*>(ar + 16 + hl * 8);
#pragma unroll
      for (int nsub = 0; nsub < 2; ++nsub)
        acc[msub][nsub] = __builtin_amdgcn_wmma_f32_16x16x32_bf16(
            false, afr.v, false, bfr[nsub].v, (short)0, acc[msub][nsub], false, false);
    }
  };

  unsigned short* buf0 = (unsigned short*)smemraw;
  unsigned short* buf1 = (unsigned short*)smemraw + 5120;

  issue_tile(0, buf0);
  for (int i2 = 0; i2 < 16; ++i2) {
    const int kk0 = i2 * 64;
    issue_tile(kk0 + 32, buf1);
    wait_async2();
    __syncthreads();
    compute(buf0, kk0);
    __syncthreads();
    if (i2 + 1 < 16) {
      issue_tile(kk0 + 64, buf0);
      wait_async2();
    } else {
      wait_async0();
    }
    __syncthreads();
    compute(buf1, kk0 + 32);
    __syncthreads();
  }
  // epilogue in two 64-column halves through LDS (fp32)
#pragma unroll
  for (int nh = 0; nh < 2; ++nh) {
    __syncthreads();
    if ((wn >> 1) == nh) {
#pragma unroll
      for (int msub = 0; msub < 4; ++msub)
#pragma unroll
        for (int nsub = 0; nsub < 2; ++nsub) {
          int cih = (wn & 1) * 32 + nsub * 16 + lc;
          float bv = bias[nbase + nh * 64 + cih];
#pragma unroll
          for (int r = 0; r < 8; ++r) {
            int mrow = wm * 64 + msub * 16 + hl * 8 + r;
            smemF[mrow * 68 + cih] = acc[msub][nsub][r] + bv;
          }
        }
    }
    __syncthreads();
    {
      int row = tid >> 1, h2 = tid & 1;
      const float4* sp = reinterpret_cast<const float4*>(smemF + row * 68 + h2 * 32);
      float4* dq = reinterpret_cast<float4*>(out + (size_t)(mbase + row) * kD + nbase + nh * 64 + h2 * 32);
#pragma unroll
      for (int i = 0; i < 8; ++i) dq[i] = sp[i];
    }
  }
}

// ---------------------------------------------------------------- launch
extern "C" void kernel_launch(void* const* d_in, const int* in_sizes, int n_in,
                              void* d_out, int out_size, void* d_ws, size_t ws_size,
                              hipStream_t stream) {
  (void)in_sizes; (void)n_in; (void)out_size; (void)ws_size;
  const float* x     = (const float*)d_in[0];
  const float* w_qkv = (const float*)d_in[1];
  const float* b_qkv = (const float*)d_in[2];
  const float* w_out = (const float*)d_in[3];
  const float* b_out = (const float*)d_in[4];
  float* outp = (float*)d_out;

  char* ws = (char*)d_ws;
  size_t off = 0;
  auto carve = [&](size_t bytes) {
    char* p = ws + off;
    off = (off + bytes + 255) & ~(size_t)255;
    return p;
  };
  unsigned short* wqkv_bf = (unsigned short*)carve((size_t)kE * kD * 2);
  unsigned short* wout_bf = (unsigned short*)carve((size_t)kD * kD * 2);
  unsigned short* xb      = (unsigned short*)carve((size_t)kB * kS * kD * 2);
  unsigned short* qb      = (unsigned short*)carve((size_t)kB * kH * kS * kHD * 2);
  unsigned short* ktb     = (unsigned short*)carve((size_t)kB * kH * kS * kHD * 2);
  unsigned short* vtb     = (unsigned short*)carve((size_t)kB * kH * kS * kHD * 2);
  unsigned short* kvb     = (unsigned short*)carve((size_t)kB * kH * kHD * kHD * 2);
  float*          invks   = (float*)carve((size_t)kB * kH * kHD * 4);
  unsigned short* attn    = (unsigned short*)carve((size_t)kB * kS * kD * 2);

  cvt_f32_bf16<<<512, 256, 0, stream>>>(w_qkv, wqkv_bf, kE * kD / 4);
  cvt_f32_bf16<<<256, 256, 0, stream>>>(w_out, wout_bf, kD * kD / 4);
  cvt_f32_bf16<<<1024, 256, 0, stream>>>(x, xb, kB * kS * kD / 4);
  gemm_qkv<<<dim3(128, 24), 256, 0, stream>>>(xb, wqkv_bf, b_qkv, qb, ktb, vtb);
  kv_state<<<64, 128, 0, stream>>>(ktb, vtb, kvb, invks);
  attn_apply<<<dim3(32, 64), 256, 0, stream>>>(qb, kvb, invks, attn);
  gemm_out<<<dim3(128, 8), 256, 0, stream>>>(attn, wout_bf, b_out, outp);
}
